// FedotpSiglip_60919816126666
// MI455X (gfx1250) — matrix operand order repaired and checked by reading, not compiled, not run on hardware
//
#include <hip/hip_runtime.h>
#include <hip/hip_fp16.h>

typedef __attribute__((ext_vector_type(16))) _Float16 v16h;
typedef __attribute__((ext_vector_type(8)))  _Float16 v8h;
typedef __attribute__((ext_vector_type(8)))  float    v8f;
typedef __attribute__((ext_vector_type(4)))  float    v4f;

#define DDIM   768
#define MROWS  196
#define IMB    256
#define NCLS   200
#define NCOLS  400                 // N * n_cls = 2*200
#define RTOT   (MROWS * IMB)       // 50176
#define NTILES (NCOLS / 16)        // 25
#define MBLKS  (RTOT / 64)         // 784
#define MGRPS  (MBLKS / 8)         // 98 (8 waves x 64 rows = 512 rows / block)
#define BPAD   776                 // 768 + 8 halves pad -> 1552B row stride (bank-safe)

static __device__ __forceinline__ v16h cat8(v8h lo, v8h hi) {
  return __builtin_shufflevector(lo, hi, 0,1,2,3,4,5,6,7,8,9,10,11,12,13,14,15);
}

// ---------------------------------------------------------------------------
// Kernel 1: row-wise L2 normalize (768 wide) + convert to f16. One wave/row.
// ---------------------------------------------------------------------------
__global__ void __launch_bounds__(256)
norm_rows_f16(const float* __restrict__ in, _Float16* __restrict__ out, int rows) {
  const int wid  = blockIdx.x * (blockDim.x >> 5) + (threadIdx.x >> 5);
  const int lane = threadIdx.x & 31;
  if (wid >= rows) return;

  const float* src = in + (size_t)wid * DDIM + lane * 24;
  float vals[24];
  float ss = 0.f;
#pragma unroll
  for (int i = 0; i < 6; ++i) {
    v4f v = ((const v4f*)src)[i];
#pragma unroll
    for (int j = 0; j < 4; ++j) { vals[4*i + j] = v[j]; ss += v[j] * v[j]; }
  }
#pragma unroll
  for (int o = 16; o > 0; o >>= 1) ss += __shfl_xor(ss, o, 32);
  const float r = rsqrtf(ss);

  v8h* dst = (v8h*)(out + (size_t)wid * DDIM + lane * 24);
#pragma unroll
  for (int ch = 0; ch < 3; ++ch) {
    v8h o;
#pragma unroll
    for (int j = 0; j < 8; ++j) o[j] = (_Float16)(vals[ch*8 + j] * r);
    dst[ch] = o;
  }
}

// ---------------------------------------------------------------------------
// Kernel 2: WMMA GEMM  sim = A(50176x768) x B^T(768x400).
// One block = one 16-col N-tile x 512 rows (8 waves x 64 rows).
// B panel (16x768 f16, 24KB) staged to LDS once via global_load_async_to_lds
// (ASYNCcnt) and shared by all 8 waves; A fragments are double-buffered in
// registers so the k+32 loads overlap the k-step WMMAs.
// ---------------------------------------------------------------------------
__global__ void __launch_bounds__(256)
wmma_gemm(const _Float16* __restrict__ A, const _Float16* __restrict__ B,
          float* __restrict__ C) {
  __shared__ _Float16 Bsh[16 * BPAD];

  const int nt   = blockIdx.x / MGRPS;       // 0..24
  const int mgrp = blockIdx.x - nt * MGRPS;  // 0..97
  const int wv   = threadIdx.x >> 5;         // 0..7
  const int lane = threadIdx.x & 31;
  const int mblk = mgrp * 8 + wv;

  // ---- stage B panel (16 x 768 halves = 1536 x 16B chunks) into LDS ----
#pragma unroll
  for (int i = 0; i < 6; ++i) {
    const int id  = threadIdx.x + 256 * i;   // 0..1535
    const int row = id / 96;                 // 96 16B-chunks per 768-half row
    const int ck  = id - row * 96;
    const _Float16* src = B + (size_t)(nt * 16 + row) * DDIM + ck * 8;
    const unsigned dst  = (unsigned)(uintptr_t)&Bsh[row * BPAD + ck * 8];
    asm volatile("global_load_async_to_lds_b128 %0, %1, off"
                 :: "v"(dst), "v"((unsigned long long)(uintptr_t)src)
                 : "memory");
  }
  asm volatile("s_wait_asynccnt 0x0" ::: "memory");
  __syncthreads();

  const int col    = nt * 16 + (lane & 15);
  const int ksel_a = (lane >> 4) * 8;    // A frag: halves0-7 = K+sel, 8-15 = K+sel+16
  const int ksel_b = (lane >> 4) * 8;    // B frag: halves0-7 = K+sel, 8-15 = K+sel+16

  const _Float16* ap0 = A + (size_t)(mblk * 64 + (lane & 15)) * DDIM + ksel_a;
  const _Float16* bls = Bsh + (lane & 15) * BPAD + ksel_b;

  const v8f zero = {0.f,0.f,0.f,0.f,0.f,0.f,0.f,0.f};
  v8f acc[4] = {zero, zero, zero, zero};

  v8h alo[2][4], ahi[2][4];              // double-buffered A fragments
#pragma unroll
  for (int i = 0; i < 4; ++i) {          // prologue: k = 0
    const _Float16* ap = ap0 + (size_t)(i * 16) * DDIM;
    alo[0][i] = *(const v8h*)ap;
    ahi[0][i] = *(const v8h*)(ap + 16);
  }

#pragma unroll
  for (int s = 0; s < 24; ++s) {         // 24 K-steps of 32
    const int buf = s & 1;
    if (s + 1 < 24) {                    // issue next-step loads first
      const int kn = (s + 1) * 32;
#pragma unroll
      for (int i = 0; i < 4; ++i) {
        const _Float16* ap = ap0 + (size_t)(i * 16) * DDIM + kn;
        alo[buf ^ 1][i] = *(const v8h*)ap;
        ahi[buf ^ 1][i] = *(const v8h*)(ap + 16);
        __builtin_prefetch(ap + 64, 0, 1);   // global_prefetch_b8, 2 steps ahead
      }
    }
    const int k = s * 32;
    v16h bf = cat8(*(const v8h*)(bls + k), *(const v8h*)(bls + k + 16));
#pragma unroll
    for (int i = 0; i < 4; ++i)
      acc[i] = __builtin_amdgcn_wmma_f32_16x16x32_f16(
          false, cat8(alo[buf][i], ahi[buf][i]), false, bf,
          (short)0, acc[i], false, false);
  }

  const int rsel = (lane >> 4) * 8;      // C/D: VGPR v -> row v (lanes<16) / v+8
#pragma unroll
  for (int i = 0; i < 4; ++i) {
    float* cp = C + (size_t)(mblk * 64 + i * 16 + rsel) * NCOLS + col;
#pragma unroll
    for (int vr = 0; vr < 8; ++vr) cp[(size_t)vr * NCOLS] = acc[i][vr];
  }
}

// ---------------------------------------------------------------------------
// Kernel 3: entropic partial-OT scaling, one (b,c) problem per wave.
// K (196x2) lives in 14 VGPRs/lane; 100 register-resident iterations with two
// butterfly reductions each; invalid rows (m>=196) carry K=0 -> contribute 0.
// ---------------------------------------------------------------------------
__global__ void __launch_bounds__(256)
sinkhorn_kernel(const float* __restrict__ sim, const float* __restrict__ logit_scale,
                float* __restrict__ out) {
  const int wid  = blockIdx.x * 8 + (threadIdx.x >> 5);   // problem id b*200+c
  const int lane = threadIdx.x & 31;
  const int b = wid / NCLS;
  const int c = wid - b * NCLS;

  float k0[7], k1[7], s0[7], s1[7], u[7];
#pragma unroll
  for (int t = 0; t < 7; ++t) {
    const int m = t * 32 + lane;
    const bool valid = (m < MROWS);
    const int mm = valid ? m : 0;
    const size_t off = ((size_t)mm * IMB + b) * NCOLS + c;   // [m][b][n][c]
    const float sa = sim[off];            // n = 0
    const float sb = sim[off + NCLS];     // n = 1
    s0[t] = sa; s1[t] = sb;
    k0[t] = valid ? __expf((sa - 1.f) * 10.f) : 0.f;   // exp(-(1-sim)/0.1)
    k1[t] = valid ? __expf((sb - 1.f) * 10.f) : 0.f;
  }

  float v0 = 1.f, v1 = 1.f;
  for (int it = 0; it < 100; ++it) {
    float a0 = 0.f, a1 = 0.f;
#pragma unroll
    for (int t = 0; t < 7; ++t) {
      const float s  = 196.f * (k0[t] * v0 + k1[t] * v1);   // (Kp @ v), Kp = M*K
      const float uu = fminf(__builtin_amdgcn_rcpf(s), 1.f);
      u[t] = uu;
      a0 += k0[t] * uu;
      a1 += k1[t] * uu;
    }
#pragma unroll
    for (int o = 16; o > 0; o >>= 1) {
      a0 += __shfl_xor(a0, o, 32);
      a1 += __shfl_xor(a1, o, 32);
    }
    v0 = 0.4f * __builtin_amdgcn_rcpf(a0);   // v = b_marg / (K^T u), b_marg=0.4
    v1 = 0.4f * __builtin_amdgcn_rcpf(a1);
  }

  float part = 0.f;
#pragma unroll
  for (int t = 0; t < 7; ++t)
    part += u[t] * (k0[t] * s0[t] * v0 + k1[t] * s1[t] * v1);   // sum T*sim
#pragma unroll
  for (int o = 16; o > 0; o >>= 1) part += __shfl_xor(part, o, 32);

  if (lane == 0) out[wid] = __expf(logit_scale[0]) * part;
}

// ---------------------------------------------------------------------------
extern "C" void kernel_launch(void* const* d_in, const int* in_sizes, int n_in,
                              void* d_out, int out_size, void* d_ws, size_t ws_size,
                              hipStream_t stream) {
  const float* imf = (const float*)d_in[0];   // (196, 256, 768) f32
  const float* txf = (const float*)d_in[1];   // (2, 200, 768)   f32
  const float* ls  = (const float*)d_in[2];   // (1,)            f32

  _Float16* A_h = (_Float16*)d_ws;                       // 50176 x 768 f16
  _Float16* B_h = A_h + (size_t)RTOT * DDIM;             //   400 x 768 f16
  float*    sim = (float*)(B_h + (size_t)NCOLS * DDIM);  // 50176 x 400 f32

  norm_rows_f16<<<RTOT  / 8, 256, 0, stream>>>(imf, A_h, RTOT);
  norm_rows_f16<<<NCOLS / 8, 256, 0, stream>>>(txf, B_h, NCOLS);
  wmma_gemm<<<NTILES * MGRPS, 256, 0, stream>>>(A_h, B_h, sim);
  sinkhorn_kernel<<<(IMB * NCLS) / 8, 256, 0, stream>>>(sim, ls, (float*)d_out);
}
